// LSTMDecoder_63977832841221
// MI455X (gfx1250) — matrix operand order repaired and checked by reading, not compiled, not run on hardware
//
#include <hip/hip_runtime.h>

// ---------------- Problem constants ----------------
#define BB   64
#define PP   196
#define EE   2048
#define DD   512
#define AA   512
#define MM   512
#define VV   10000
#define TT   31
#define TMAX 30
#define KC   64                             // K chunk (all K here are %64==0)
#define VPAD 10112                          // ceil(10000/128)*128
#define PRED_ELEMS ((long)BB * TMAX * VV)   // 19,200,000

// ---------------- WMMA types (gfx1250, wave32) ----------------
typedef __attribute__((ext_vector_type(16))) __bf16 v16bf;
typedef __attribute__((ext_vector_type(8)))  float  v8f;

union BFrag { v16bf v; unsigned short s[16]; };
union AccF  { v8f   v; float f[8]; };

__device__ __forceinline__ unsigned short f2bf(float x) {
  unsigned u = __float_as_uint(x);
  u += 0x7FFFu + ((u >> 16) & 1u);          // round-to-nearest-even
  return (unsigned short)(u >> 16);
}
__device__ __forceinline__ float sigmoidf(float x) { return 1.0f / (1.0f + __expf(-x)); }

// ---------------------------------------------------------------------------
// One-time: W (N x K, f32, row-major) -> Wt (K x Npad, bf16), zero-padded in n.
// LDS-tiled transpose, coalesced on both sides. K % 32 == 0, Npad % 32 == 0.
// ---------------------------------------------------------------------------
__global__ void __launch_bounds__(256)
transpose_bf16_kernel(const float* __restrict__ W, unsigned short* __restrict__ Wt,
                      int N, int K, int Npad)
{
  __shared__ float tile[32][33];
  const int k0 = blockIdx.x * 32;
  const int n0 = blockIdx.y * 32;
  const int tx = threadIdx.x & 31;
  const int ty = threadIdx.x >> 5;          // 0..7
#pragma unroll
  for (int i = 0; i < 4; ++i) {
    const int n = n0 + ty + 8 * i;
    tile[ty + 8 * i][tx] = (n < N) ? W[(long)n * K + k0 + tx] : 0.0f;
  }
  __syncthreads();
#pragma unroll
  for (int i = 0; i < 4; ++i) {
    const int k = k0 + ty + 8 * i;
    Wt[(long)k * Npad + n0 + tx] = f2bf(tile[tx][ty + 8 * i]);
  }
}

// ---------------------------------------------------------------------------
// GEMM:  C[m, n] = rowscale[m] * ( A[rowmap(m), :K] @ Bt[:, n]  (Bt = K x Npad bf16)
//                                   + bias[n] + (addC ? C_old : 0) )
// A f32 row-major (lda=K), converted to bf16 in staging; Bt pre-converted,
// pre-transposed, zero-padded -> branch-free 128-bit copies into LDS.
// Block = 256 threads = 8 waves; tile 32(M) x 128(N); wave grid 2x4,
// 2 accumulators/wave; K chunk 64 -> 4 wmmas/wave per barrier;
// double-buffered LDS, 1 barrier per K-chunk.
// Requires: M % 32 == 0, K % 64 == 0, ldbt % 8 == 0, grid.x covers Npad/128.
// ---------------------------------------------------------------------------
__global__ void __launch_bounds__(256)
gemm_bf16_wmma(const float* __restrict__ A, const int* __restrict__ rowmap,
               const unsigned short* __restrict__ Bt, int ldbt,
               const float* __restrict__ bias, const float* __restrict__ rowscale,
               float* __restrict__ C, int N, int K, long ldc, int addC)
{
  __shared__ alignas(16) unsigned short lA[2][32][72];    // [buf][m][k]  (64k + pad)
  __shared__ alignas(16) unsigned short lB[2][64][136];   // [buf][k][n]  (128n + pad)

  const int tid  = threadIdx.x;
  const int wave = tid >> 5;
  const int lane = tid & 31;
  const int wm   = wave >> 2;                 // 0..1  (M strip)
  const int wn   = wave & 3;                  // 0..3  (N strip)
  const int m0   = blockIdx.y * 32;
  const int n0b  = blockIdx.x * 128;

  // ---- A staging: 32 rows x 64 k, two float4 per thread ----
  const int ar  = tid >> 3;                   // 0..31
  const int akk = (tid & 7) * 4;              // 0,4,..28  (second quad at +32)
  const long arow = rowmap ? (long)rowmap[m0 + ar] : (long)(m0 + ar);
  const float* aptr = A + arow * (long)K + akk;

  // ---- B staging: 64 k x 128 n bf16, four 16-byte granules per thread ----
  const int bkk = tid >> 4;                   // 0..15 (strided +16 per granule)
  const int bnc = (tid & 15) * 8;             // 0..120
  const unsigned short* bsrc = Bt + (long)bkk * ldbt + n0b + bnc;
  const long bstep16 = (long)16 * ldbt;       // 16 k-rows

  v8f acc0 = {}, acc1 = {};
  float4 ra0, ra1;
  uint4  rb0, rb1, rb2, rb3;

  auto stage = [&](int buf) {
    lA[buf][ar][akk +  0] = f2bf(ra0.x);
    lA[buf][ar][akk +  1] = f2bf(ra0.y);
    lA[buf][ar][akk +  2] = f2bf(ra0.z);
    lA[buf][ar][akk +  3] = f2bf(ra0.w);
    lA[buf][ar][akk + 32] = f2bf(ra1.x);
    lA[buf][ar][akk + 33] = f2bf(ra1.y);
    lA[buf][ar][akk + 34] = f2bf(ra1.z);
    lA[buf][ar][akk + 35] = f2bf(ra1.w);
    *(uint4*)&lB[buf][bkk +  0][bnc] = rb0;   // raw 128-bit copies, no convert
    *(uint4*)&lB[buf][bkk + 16][bnc] = rb1;
    *(uint4*)&lB[buf][bkk + 32][bnc] = rb2;
    *(uint4*)&lB[buf][bkk + 48][bnc] = rb3;
  };
  auto gload = [&](long k) {
    const long boff = k * ldbt;
    ra0 = *(const float4*)(aptr + k);
    ra1 = *(const float4*)(aptr + k + 32);
    rb0 = *(const uint4*)(bsrc + boff);
    rb1 = *(const uint4*)(bsrc + boff + bstep16);
    rb2 = *(const uint4*)(bsrc + boff + 2 * bstep16);
    rb3 = *(const uint4*)(bsrc + boff + 3 * bstep16);
  };

  // ---- prologue: chunk 0 -> LDS buf 0 ----
  gload(0);
  stage(0);
  __syncthreads();

  const int row = wm * 16 + (lane & 15);
  const int kb  = (lane >> 4) * 8;             // lanes 16-31: K=8..15 / 24..31
  const int nc0 = wn * 32;

  const int nsteps = K / KC;
  for (int i = 0; i < nsteps; ++i) {
    const int cur = i & 1;
    if (i + 1 < nsteps) {
      gload((long)(i + 1) * KC);
      stage(cur ^ 1);                          // write the buffer not read now
    }
    if (i + 2 < nsteps) {                      // gfx1250 global_prefetch path
      __builtin_prefetch(aptr + (long)(i + 2) * KC, 0, 1);
      __builtin_prefetch(bsrc + (long)(i + 2) * KC * ldbt, 0, 1);
    }

    // ---- two 16x16x32 sub-steps per chunk (ISA wave32 layouts) ----
#pragma unroll
    for (int j = 0; j < 2; ++j) {
      BFrag af, bf0r, bf1r;
      const int ko = j * 32;
#pragma unroll
      for (int e = 0; e < 8; ++e)  af.s[e] = lA[cur][row][ko + kb + e];
#pragma unroll
      for (int e = 8; e < 16; ++e) af.s[e] = lA[cur][row][ko + kb + e + 8];
#pragma unroll
      for (int e = 0; e < 16; ++e) bf0r.s[e] = lB[cur][ko + kb + e + (e & 8)][nc0 + (lane & 15)];
#pragma unroll
      for (int e = 0; e < 16; ++e) bf1r.s[e] = lB[cur][ko + kb + e + (e & 8)][nc0 + 16 + (lane & 15)];

      acc0 = __builtin_amdgcn_wmma_f32_16x16x32_bf16(false, af.v, false, bf0r.v,
                                                     (short)0, acc0, false, false);
      acc1 = __builtin_amdgcn_wmma_f32_16x16x32_bf16(false, af.v, false, bf1r.v,
                                                     (short)0, acc1, false, false);
    }
    __syncthreads();
  }

  // ---- epilogue: C/D layout — VGPR r -> M = r (+8 for lanes>=16), lane -> N
  AccF r0, r1; r0.v = acc0; r1.v = acc1;
  const int mb    = m0 + wm * 16 + ((lane >> 4) << 3);
  const int nbase = n0b + wn * 32 + (lane & 15);
#pragma unroll
  for (int half = 0; half < 2; ++half) {
    const int nn = nbase + half * 16;
    const float* rf = half ? r1.f : r0.f;
    if (nn < N) {
      const float bv = bias ? bias[nn] : 0.0f;
#pragma unroll
      for (int i = 0; i < 8; ++i) {
        const long off = (long)(mb + i) * ldc + nn;
        float v = rf[i] + bv;
        if (addC)     v += C[off];
        if (rowscale) v *= rowscale[mb + i];
        C[off] = v;
      }
    }
  }
}

// ---------------------------------------------------------------------------
// Setup kernels
// ---------------------------------------------------------------------------
__global__ void sort_lens_kernel(const int* __restrict__ lens,
                                 int* __restrict__ idx, int* __restrict__ declen)
{
  __shared__ int sl[BB];
  __shared__ int sidx[BB];
  const int i = threadIdx.x;
  sl[i] = lens[i];
  __syncthreads();
  const int li = sl[i];
  int rank = 0;
  for (int j = 0; j < BB; ++j) {
    const int lj = sl[j];
    rank += (lj > li) || (lj == li && j < i);   // stable descending
  }
  sidx[rank] = i;
  __syncthreads();
  idx[i]    = sidx[i];
  declen[i] = sl[sidx[i]] - 1;
}

__global__ void rowmap_kernel(const int* __restrict__ idx, int* __restrict__ rowmap)
{
  const int r = blockIdx.x * 256 + threadIdx.x;
  if (r < BB * PP) rowmap[r] = idx[r / PP] * PP + (r % PP);
}

__global__ void mean_enc_kernel(const float* __restrict__ enc,
                                const int* __restrict__ idx,
                                float* __restrict__ mean)
{
  const int b = blockIdx.y;
  const int e = blockIdx.x * 256 + threadIdx.x;
  const float* base = enc + (long)idx[b] * PP * EE + e;
  float s = 0.0f;
  for (int p = 0; p < PP; ++p) s += base[(long)p * EE];
  mean[(long)b * EE + e] = s * (1.0f / (float)PP);
}

// ---------------------------------------------------------------------------
// Per-step kernels
// ---------------------------------------------------------------------------
__global__ void __launch_bounds__(256)
scores_softmax_kernel(const float* __restrict__ att1, const float* __restrict__ att2,
                      const float* __restrict__ wf, const float* __restrict__ bf0,
                      const int* __restrict__ declen,
                      float* __restrict__ alpha_ws, float* __restrict__ out_alpha,
                      int t)
{
  const int b    = blockIdx.x;
  const int tid  = threadIdx.x;
  const int wave = tid >> 5;
  const int lane = tid & 31;
  __shared__ float s_a2[AA];
  __shared__ float s_wf[AA];
  __shared__ float s_sc[PP + 4];
  __shared__ float red[256];

  for (int i = tid; i < AA; i += 256) {
    s_a2[i] = att2[(long)b * AA + i];
    s_wf[i] = wf[i];
  }
  __syncthreads();

  const float bias0 = bf0[0];
  for (int p = wave; p < PP; p += 8) {
    const float* a1 = att1 + ((long)b * PP + p) * AA;
    float part = 0.0f;
    for (int a = lane; a < AA; a += 32)
      part += tanhf(a1[a] + s_a2[a]) * s_wf[a];
#pragma unroll
    for (int off = 16; off > 0; off >>= 1)
      part += __shfl_xor(part, off, 32);
    if (lane == 0) s_sc[p] = part + bias0;
  }
  __syncthreads();

  const float score = (tid < PP) ? s_sc[tid] : -1e30f;
  red[tid] = score;
  __syncthreads();
  for (int s = 128; s > 0; s >>= 1) {
    if (tid < s) red[tid] = fmaxf(red[tid], red[tid + s]);
    __syncthreads();
  }
  const float mx = red[0];
  __syncthreads();
  const float ev = (tid < PP) ? __expf(score - mx) : 0.0f;
  red[tid] = ev;
  __syncthreads();
  for (int s = 128; s > 0; s >>= 1) {
    if (tid < s) red[tid] += red[tid + s];
    __syncthreads();
  }
  const float inv = 1.0f / red[0];
  if (tid < PP) {
    const float al = ev * inv;
    alpha_ws[(long)b * PP + tid] = al;                       // raw alpha for ctx
    const int active = t < declen[b];
    out_alpha[((long)b * TMAX + t) * PP + tid] = active ? al : 0.0f;
  }
}

__global__ void ctx_kernel(const float* __restrict__ enc, const int* __restrict__ idx,
                           const float* __restrict__ alpha, float* __restrict__ ctx)
{
  const int b = blockIdx.y;
  const int e = blockIdx.x * 256 + threadIdx.x;
  __shared__ float s_al[PP];
  if (threadIdx.x < PP) s_al[threadIdx.x] = alpha[(long)b * PP + threadIdx.x];
  __syncthreads();
  const float* base = enc + (long)idx[b] * PP * EE + e;
  float acc = 0.0f;
  for (int p = 0; p < PP; ++p) acc += s_al[p] * base[(long)p * EE];
  ctx[(long)b * EE + e] = acc;
}

__global__ void pack_x_kernel(const float* __restrict__ W_emb,
                              const int* __restrict__ captions,
                              const int* __restrict__ idx,
                              const float* __restrict__ ctx,
                              float* __restrict__ xcat, int t)
{
  const int b = blockIdx.y;
  const int i = blockIdx.x * 256 + threadIdx.x;   // 0..2559
  float v;
  if (i < MM) {
    const int tok = captions[idx[b] * TT + t];
    v = W_emb[(long)tok * MM + i];
  } else {
    v = ctx[(long)b * EE + (i - MM)];
  }
  xcat[(long)b * (MM + EE) + i] = v;
}

__global__ void lstm_cell_kernel(const float* __restrict__ gates,
                                 const int* __restrict__ declen,
                                 float* __restrict__ h, float* __restrict__ c,
                                 float* __restrict__ actmask, int t)
{
  const int b = blockIdx.y;
  const int d = blockIdx.x * 256 + threadIdx.x;   // 0..511
  const long gb = (long)b * (4 * DD);
  const float ig = gates[gb + d];
  const float fg = gates[gb + DD + d];
  const float gg = gates[gb + 2 * DD + d];
  const float og = gates[gb + 3 * DD + d];
  const float cc = c[(long)b * DD + d];
  const float ct = sigmoidf(fg) * cc + sigmoidf(ig) * tanhf(gg);
  const float ht = sigmoidf(og) * tanhf(ct);
  const int active = t < declen[b];
  if (active) {
    h[(long)b * DD + d] = ht;
    c[(long)b * DD + d] = ct;
  }
  if (d == 0) actmask[b] = active ? 1.0f : 0.0f;
}

// ---------------------------------------------------------------------------
// Host launcher
// ---------------------------------------------------------------------------
extern "C" void kernel_launch(void* const* d_in, const int* in_sizes, int n_in,
                              void* d_out, int out_size, void* d_ws, size_t ws_size,
                              hipStream_t stream)
{
  const float* encoder_out = (const float*)d_in[0];
  const int*   captions    = (const int*)  d_in[1];
  const int*   cap_lens    = (const int*)  d_in[2];
  const float* W_emb = (const float*)d_in[3];
  const float* We    = (const float*)d_in[4];
  const float* be    = (const float*)d_in[5];
  const float* Wd    = (const float*)d_in[6];
  const float* bd    = (const float*)d_in[7];
  const float* wf    = (const float*)d_in[8];
  const float* bf0   = (const float*)d_in[9];
  const float* Wih   = (const float*)d_in[10];
  const float* bih   = (const float*)d_in[11];
  const float* Whh   = (const float*)d_in[12];
  const float* bhh   = (const float*)d_in[13];
  const float* Wfc   = (const float*)d_in[14];
  const float* bfc   = (const float*)d_in[15];
  const float* Wh0   = (const float*)d_in[16];
  const float* bh0   = (const float*)d_in[17];
  const float* Wc0   = (const float*)d_in[18];
  const float* bc0   = (const float*)d_in[19];

  float* out_pred  = (float*)d_out;
  float* out_alpha = (float*)d_out + PRED_ELEMS;

  // ---- workspace partition ----
  float* W = (float*)d_ws;
  size_t o = 0;
  int* idx_ws    = (int*)(W + o); o += 64;
  int* declen_ws = (int*)(W + o); o += 64;
  int* rowmap_ws = (int*)(W + o); o += 12608;           // 12544 + pad
  float* actmask = W + o; o += 64;
  float* mean_e  = W + o; o += (size_t)BB * EE;
  float* h_ws    = W + o; o += (size_t)BB * DD;
  float* c_ws    = W + o; o += (size_t)BB * DD;
  float* att2_ws = W + o; o += (size_t)BB * AA;
  float* alpha_w = W + o; o += (size_t)BB * PP + 64;
  float* ctx_ws  = W + o; o += (size_t)BB * EE;
  float* xcat_ws = W + o; o += (size_t)BB * (MM + EE);
  float* gates_w = W + o; o += (size_t)BB * 4 * DD;
  float* att1_ws = W + o; o += (size_t)BB * PP * AA;    // ~25.7 MB

  // bf16 transposed weights (K x Npad each), ~29.8 MB total
  unsigned short* U = (unsigned short*)(W + o);
  size_t u = 0;
  unsigned short* WeT  = U + u; u += (size_t)EE * AA;          // 2048 x 512
  unsigned short* WdT  = U + u; u += (size_t)DD * AA;          // 512  x 512
  unsigned short* WihT = U + u; u += (size_t)(MM + EE) * 4*DD; // 2560 x 2048
  unsigned short* WhhT = U + u; u += (size_t)DD * 4*DD;        // 512  x 2048
  unsigned short* WfcT = U + u; u += (size_t)DD * VPAD;        // 512  x 10112
  unsigned short* Wh0T = U + u; u += (size_t)EE * DD;          // 2048 x 512
  unsigned short* Wc0T = U + u; u += (size_t)EE * DD;          // 2048 x 512

  // ---- one-time weight convert+transpose (f32 NxK -> bf16 KxNpad) ----
  transpose_bf16_kernel<<<dim3(EE/32, AA/32),   256, 0, stream>>>(We,  WeT,  AA,   EE, AA);
  transpose_bf16_kernel<<<dim3(DD/32, AA/32),   256, 0, stream>>>(Wd,  WdT,  AA,   DD, AA);
  transpose_bf16_kernel<<<dim3((MM+EE)/32, (4*DD)/32), 256, 0, stream>>>(Wih, WihT, 4*DD, MM+EE, 4*DD);
  transpose_bf16_kernel<<<dim3(DD/32, (4*DD)/32), 256, 0, stream>>>(Whh, WhhT, 4*DD, DD, 4*DD);
  transpose_bf16_kernel<<<dim3(DD/32, VPAD/32), 256, 0, stream>>>(Wfc, WfcT, VV,   DD, VPAD);
  transpose_bf16_kernel<<<dim3(EE/32, DD/32),   256, 0, stream>>>(Wh0, Wh0T, DD,   EE, DD);
  transpose_bf16_kernel<<<dim3(EE/32, DD/32),   256, 0, stream>>>(Wc0, Wc0T, DD,   EE, DD);

  // ---- setup ----
  sort_lens_kernel<<<1, BB, 0, stream>>>(cap_lens, idx_ws, declen_ws);
  rowmap_kernel<<<(BB * PP + 255) / 256, 256, 0, stream>>>(idx_ws, rowmap_ws);
  mean_enc_kernel<<<dim3(EE / 256, BB), 256, 0, stream>>>(encoder_out, idx_ws, mean_e);

  // h0 = mean @ Wh0^T + bh0 ; c0 = mean @ Wc0^T + bc0
  gemm_bf16_wmma<<<dim3(DD / 128, BB / 32), 256, 0, stream>>>(
      mean_e, nullptr, Wh0T, DD, bh0, nullptr, h_ws, DD, EE, DD, 0);
  gemm_bf16_wmma<<<dim3(DD / 128, BB / 32), 256, 0, stream>>>(
      mean_e, nullptr, Wc0T, DD, bc0, nullptr, c_ws, DD, EE, DD, 0);

  // att1 = enc_sorted @ We^T + be   (12544 x 512, K=2048)
  gemm_bf16_wmma<<<dim3(AA / 128, (BB * PP) / 32), 256, 0, stream>>>(
      encoder_out, rowmap_ws, WeT, AA, be, nullptr, att1_ws, AA, EE, AA, 0);

  // ---- recurrence ----
  for (int t = 0; t < TMAX; ++t) {
    gemm_bf16_wmma<<<dim3(AA / 128, BB / 32), 256, 0, stream>>>(
        h_ws, nullptr, WdT, AA, bd, nullptr, att2_ws, AA, DD, AA, 0);
    scores_softmax_kernel<<<BB, 256, 0, stream>>>(
        att1_ws, att2_ws, wf, bf0, declen_ws, alpha_w, out_alpha, t);
    ctx_kernel<<<dim3(EE / 256, BB), 256, 0, stream>>>(
        encoder_out, idx_ws, alpha_w, ctx_ws);
    pack_x_kernel<<<dim3((MM + EE) / 256, BB), 256, 0, stream>>>(
        W_emb, captions, idx_ws, ctx_ws, xcat_ws, t);
    gemm_bf16_wmma<<<dim3((4 * DD) / 128, BB / 32), 256, 0, stream>>>(
        xcat_ws, nullptr, WihT, 4 * DD, bih, nullptr, gates_w, 4 * DD, MM + EE, 4 * DD, 0);
    gemm_bf16_wmma<<<dim3((4 * DD) / 128, BB / 32), 256, 0, stream>>>(
        h_ws, nullptr, WhhT, 4 * DD, bhh, nullptr, gates_w, 4 * DD, DD, 4 * DD, 1);
    lstm_cell_kernel<<<dim3(DD / 256, BB), 256, 0, stream>>>(
        gates_w, declen_ws, h_ws, c_ws, actmask, t);
    gemm_bf16_wmma<<<dim3(VPAD / 128, BB / 32), 256, 0, stream>>>(
        h_ws, nullptr, WfcT, VPAD, bfc, actmask, out_pred + (long)t * VV,
        VV, DD, (long)TMAX * VV, 0);
  }
}
